// TwoHeadLSTM_22969485099268
// MI455X (gfx1250) — matrix operand order repaired and checked by reading, not compiled, not run on hardware
//
#include <hip/hip_runtime.h>

#define T_STEPS 256
#define BATCH   32
#define HID     1024
#define EMB     128
#define NEMB    25
#define LINW    1024
#define NWG     64          // workgroups per branch
#define UPW     16          // hidden units per WG (HID / NWG)
#define THREADS 128         // 4 waves: wave g computes gate block g

typedef __bf16 bf16_t;
typedef __attribute__((ext_vector_type(8)))  float  v8f;
typedef __attribute__((ext_vector_type(8)))  bf16_t v8bf;
typedef __attribute__((ext_vector_type(16))) bf16_t v16bf;

static __device__ __forceinline__ unsigned short f2bf(float f) {
  unsigned u = __float_as_uint(f);
  unsigned r = u + 0x7FFFu + ((u >> 16) & 1u);   // round-to-nearest-even
  return (unsigned short)(r >> 16);
}

static __device__ __forceinline__ float sigm(float x) {
  return 1.0f / (1.0f + __expf(-x));
}

union V16U { v16bf v; v8bf h[2]; };

// A-fragment (16x32, rows m0..m0+15, cols k0..k0+31) from row-major bf16, ld elems.
// ISA layout: lanes 0-15 hold K[0..8)+K[16..24), lanes 16-31 hold K[8..16)+K[24..32).
static __device__ __forceinline__ v16bf load_afrag(const unsigned short* base,
                                                   int m0, int k0, int ld, int lane) {
  int m  = m0 + (lane & 15);
  int kb = k0 + ((lane >> 4) << 3);
  const unsigned short* p = base + m * ld + kb;
  V16U u;
  u.h[0] = *(const v8bf*)(p);
  u.h[1] = *(const v8bf*)(p + 16);
  return u.v;
}

// B-fragment (32x16 = W^T tile): rows n0..n0+15 of row-major W, cols k0..k0+31.
// ISA layout: lanes 0-15 hold K[0..8)+K[16..24), lanes 16-31 hold K[8..16)+K[24..32).
static __device__ __forceinline__ v16bf load_bfrag(const unsigned short* wbase,
                                                   int n0, int k0, int ld, int lane) {
  const unsigned short* p = wbase + (n0 + (lane & 15)) * ld + k0 + ((lane >> 4) << 3);
  V16U u;
  u.h[0] = *(const v8bf*)(p);
  u.h[1] = *(const v8bf*)(p + 16);
  return u.v;
}

#define WMMA_BF16(A, B, C) \
  __builtin_amdgcn_wmma_f32_16x16x32_bf16(false, (A), false, (B), (short)0, (C), false, false)

// ---------------------------------------------------------------- converts
__global__ void cvt_bf16_kernel(const float* __restrict__ src,
                                unsigned short* __restrict__ dst, int n) {
  int i = blockIdx.x * blockDim.x + threadIdx.x;
  int stride = gridDim.x * blockDim.x;
  for (; i < n; i += stride) dst[i] = f2bf(src[i]);
}

// ---------------------------------------------------------------- grid sync
static __device__ __forceinline__ void grid_sync(unsigned* ctr, unsigned target) {
  __syncthreads();
  if (threadIdx.x == 0) {
    __threadfence();
    __hip_atomic_fetch_add(ctr, 1u, __ATOMIC_RELEASE, __HIP_MEMORY_SCOPE_AGENT);
    while (__hip_atomic_load(ctr, __ATOMIC_ACQUIRE, __HIP_MEMORY_SCOPE_AGENT) < target) {
      __builtin_amdgcn_s_sleep(1);
    }
  }
  __syncthreads();
  __threadfence();   // acquire: make peer WG stores visible to all lanes
}

// ---------------------------------------------------------------- recurrent
__global__ __launch_bounds__(THREADS, 1)
void lstm_recurrent(const float* __restrict__ emb,
                    const unsigned short* __restrict__ Wih0,   // [4H][E]  bf16
                    const unsigned short* __restrict__ Whh0,   // [4H][H]  bf16
                    const unsigned short* __restrict__ Wih1,   // [4H][H]  bf16
                    const unsigned short* __restrict__ Whh1,   // [4H][H]  bf16
                    const float* __restrict__ bih0, const float* __restrict__ bhh0,
                    const float* __restrict__ bih1, const float* __restrict__ bhh1,
                    const int* __restrict__ tokL, const int* __restrict__ tokR,
                    const int* __restrict__ lenL, const int* __restrict__ lenR,
                    unsigned short* __restrict__ h1g,   // [2 branch][2 buf][B][H] bf16
                    unsigned short* __restrict__ h2g,
                    unsigned* __restrict__ sync_base) {
  const int branch = blockIdx.x / NWG;
  const int wg     = blockIdx.x % NWG;
  const int n0     = wg * UPW;                 // hidden-unit slice base
  const int wave   = threadIdx.x >> 5;         // gate id (i,f,g,o)
  const int lane   = threadIdx.x & 31;

  const int* tok = branch ? tokR : tokL;
  const int* len = branch ? lenR : lenL;
  unsigned short* h1b = h1g + branch * (2 * BATCH * HID);
  unsigned short* h2b = h2g + branch * (2 * BATCH * HID);
  unsigned* ctr = sync_base + branch * 32;     // separate cache lines per branch

  __shared__ float          semb[NEMB * EMB];        // 12.8 KB
  __shared__ unsigned short sx[BATCH * EMB];         // 8 KB   x[t] tile (bf16)
  __shared__ float          stile[4][BATCH * UPW];   // 8 KB   per-gate output tiles
  __shared__ float          sc1[BATCH * UPW];        // 2 KB   cell state L0 slice
  __shared__ float          sc2[BATCH * UPW];        // 2 KB   cell state L1 slice
  __shared__ int            slen[BATCH];

  for (int i = threadIdx.x; i < NEMB * EMB; i += THREADS) semb[i] = emb[i];
  for (int i = threadIdx.x; i < BATCH * UPW; i += THREADS) { sc1[i] = 0.f; sc2[i] = 0.f; }
  if (threadIdx.x < BATCH) slen[threadIdx.x] = len[threadIdx.x];
  __syncthreads();

  // Per-thread gate column is invariant: hoist bias sums into registers.
  const int ug = n0 + (threadIdx.x & 15);
  float bs0[4], bs1[4];
  #pragma unroll
  for (int g = 0; g < 4; ++g) {
    bs0[g] = bih0[g * HID + ug] + bhh0[g * HID + ug];
    bs1[g] = bih1[g * HID + ug] + bhh1[g * HID + ug];
  }

  // Warm L2 with this WG's weight slices (global_prefetch_b8).
  __builtin_prefetch((const void*)(Whh0 + (wave * HID + n0) * HID), 0, 0);
  __builtin_prefetch((const void*)(Wih1 + (wave * HID + n0) * HID), 0, 0);
  __builtin_prefetch((const void*)(Whh1 + (wave * HID + n0) * HID), 0, 0);

  const unsigned short* W0h = Whh0 + (wave * HID + n0) * HID;
  const unsigned short* W0x = Wih0 + (wave * HID + n0) * EMB;
  const unsigned short* W1i = Wih1 + (wave * HID + n0) * HID;
  const unsigned short* W1h = Whh1 + (wave * HID + n0) * HID;

  unsigned sync_round = 0;

  for (int t = 0; t < T_STEPS; ++t) {
    const int p = t & 1, q = 1 - p;            // ping-pong state buffers
    const unsigned short* h1p = h1b + p * (BATCH * HID);
    unsigned short*       h1q = h1b + q * (BATCH * HID);
    const unsigned short* h2p = h2b + p * (BATCH * HID);
    unsigned short*       h2q = h2b + q * (BATCH * HID);

    // ---- stage x[t] = emb[token] (or 0) as bf16 tile [B][E] in LDS
    {
      int b  = threadIdx.x >> 2;
      int e0 = (threadIdx.x & 3) << 5;
      int tk = tok[t * BATCH + b];
      if (tk >= 0) {
        const float* er = semb + tk * EMB;
        #pragma unroll
        for (int e = 0; e < 32; ++e) sx[b * EMB + e0 + e] = f2bf(er[e0 + e]);
      } else {
        #pragma unroll
        for (int e = 0; e < 32; ++e) sx[b * EMB + e0 + e] = 0;
      }
    }
    __syncthreads();

    // ---- layer 0 gates: wave computes [32 x 16] tile of gate block `wave`
    {
      v8f acc0 = {}; v8f acc1 = {};
      // depth-2 software pipeline over K (h-part)
      v16bf b0 = load_bfrag(W0h, 0, 0,  HID, lane);
      v16bf p0 = load_afrag(h1p, 0, 0,  HID, lane);
      v16bf p1 = load_afrag(h1p, 16, 0, HID, lane);
      v16bf b1 = load_bfrag(W0h, 0, 32,  HID, lane);
      v16bf q0 = load_afrag(h1p, 0, 32,  HID, lane);
      v16bf q1 = load_afrag(h1p, 16, 32, HID, lane);
      #pragma unroll 2
      for (int k0 = 0; k0 < HID - 64; k0 += 32) {
        acc0 = WMMA_BF16(p0, b0, acc0);
        acc1 = WMMA_BF16(p1, b0, acc1);
        b0 = b1; p0 = q0; p1 = q1;
        b1 = load_bfrag(W0h, 0, k0 + 64,  HID, lane);
        q0 = load_afrag(h1p, 0, k0 + 64,  HID, lane);
        q1 = load_afrag(h1p, 16, k0 + 64, HID, lane);
      }
      acc0 = WMMA_BF16(p0, b0, acc0);
      acc1 = WMMA_BF16(p1, b0, acc1);
      acc0 = WMMA_BF16(q0, b1, acc0);
      acc1 = WMMA_BF16(q1, b1, acc1);
      // x-part: 4 iterations, loads hoisted into one clause by full unroll
      #pragma unroll
      for (int k0 = 0; k0 < EMB; k0 += 32) {
        v16bf bf = load_bfrag(W0x, 0, k0, EMB, lane);
        v16bf a0 = load_afrag(sx, 0,  k0, EMB, lane);
        v16bf a1 = load_afrag(sx, 16, k0, EMB, lane);
        acc0 = WMMA_BF16(a0, bf, acc0);
        acc1 = WMMA_BF16(a1, bf, acc1);
      }
      int nn = lane & 15, mb = (lane >> 4) << 3;
      #pragma unroll
      for (int r = 0; r < 8; ++r) {
        stile[wave][(mb + r) * UPW + nn]      = acc0[r];
        stile[wave][(16 + mb + r) * UPW + nn] = acc1[r];
      }
    }
    __syncthreads();

    // ---- layer 0 elementwise (i,f,g,o -> h1,c1), packed-seq freeze via mask
    #pragma unroll
    for (int j = 0; j < (BATCH * UPW) / THREADS; ++j) {
      int idx = threadIdx.x + j * THREADS;
      int b = idx >> 4;
      float ig = stile[0][idx] + bs0[0];
      float fg = stile[1][idx] + bs0[1];
      float gg = stile[2][idx] + bs0[2];
      float og = stile[3][idx] + bs0[3];
      unsigned short newv = h1p[b * HID + ug];
      if (t < slen[b]) {
        float cn = sigm(fg) * sc1[idx] + sigm(ig) * tanhf(gg);
        sc1[idx] = cn;
        newv = f2bf(sigm(og) * tanhf(cn));
      }
      h1q[b * HID + ug] = newv;
    }
    grid_sync(ctr, NWG * (++sync_round));      // h1q complete across branch

    // ---- layer 1 gates: W_ih1*h1_new + W_hh1*h2_prev, merged K loop
    {
      v8f acc0 = {}; v8f acc1 = {};
      v16bf bi0 = load_bfrag(W1i, 0, 0,  HID, lane);
      v16bf bh0 = load_bfrag(W1h, 0, 0,  HID, lane);
      v16bf x0  = load_afrag(h1q, 0, 0,  HID, lane);
      v16bf x1  = load_afrag(h1q, 16, 0, HID, lane);
      v16bf y0  = load_afrag(h2p, 0, 0,  HID, lane);
      v16bf y1  = load_afrag(h2p, 16, 0, HID, lane);
      v16bf bi1 = load_bfrag(W1i, 0, 32,  HID, lane);
      v16bf bh1 = load_bfrag(W1h, 0, 32,  HID, lane);
      v16bf u0  = load_afrag(h1q, 0, 32,  HID, lane);
      v16bf u1  = load_afrag(h1q, 16, 32, HID, lane);
      v16bf w0  = load_afrag(h2p, 0, 32,  HID, lane);
      v16bf w1  = load_afrag(h2p, 16, 32, HID, lane);
      #pragma unroll 2
      for (int k0 = 0; k0 < HID - 64; k0 += 32) {
        acc0 = WMMA_BF16(x0, bi0, acc0);
        acc1 = WMMA_BF16(x1, bi0, acc1);
        acc0 = WMMA_BF16(y0, bh0, acc0);
        acc1 = WMMA_BF16(y1, bh0, acc1);
        bi0 = bi1; bh0 = bh1; x0 = u0; x1 = u1; y0 = w0; y1 = w1;
        bi1 = load_bfrag(W1i, 0, k0 + 64,  HID, lane);
        bh1 = load_bfrag(W1h, 0, k0 + 64,  HID, lane);
        u0  = load_afrag(h1q, 0, k0 + 64,  HID, lane);
        u1  = load_afrag(h1q, 16, k0 + 64, HID, lane);
        w0  = load_afrag(h2p, 0, k0 + 64,  HID, lane);
        w1  = load_afrag(h2p, 16, k0 + 64, HID, lane);
      }
      acc0 = WMMA_BF16(x0, bi0, acc0);
      acc1 = WMMA_BF16(x1, bi0, acc1);
      acc0 = WMMA_BF16(y0, bh0, acc0);
      acc1 = WMMA_BF16(y1, bh0, acc1);
      acc0 = WMMA_BF16(u0, bi1, acc0);
      acc1 = WMMA_BF16(u1, bi1, acc1);
      acc0 = WMMA_BF16(w0, bh1, acc0);
      acc1 = WMMA_BF16(w1, bh1, acc1);
      int nn = lane & 15, mb = (lane >> 4) << 3;
      #pragma unroll
      for (int r = 0; r < 8; ++r) {
        stile[wave][(mb + r) * UPW + nn]      = acc0[r];
        stile[wave][(16 + mb + r) * UPW + nn] = acc1[r];
      }
    }
    __syncthreads();

    // ---- layer 1 elementwise -> h2, c2
    #pragma unroll
    for (int j = 0; j < (BATCH * UPW) / THREADS; ++j) {
      int idx = threadIdx.x + j * THREADS;
      int b = idx >> 4;
      float ig = stile[0][idx] + bs1[0];
      float fg = stile[1][idx] + bs1[1];
      float gg = stile[2][idx] + bs1[2];
      float og = stile[3][idx] + bs1[3];
      unsigned short newv = h2p[b * HID + ug];
      if (t < slen[b]) {
        float cn = sigm(fg) * sc2[idx] + sigm(ig) * tanhf(gg);
        sc2[idx] = cn;
        newv = f2bf(sigm(og) * tanhf(cn));
      }
      h2q[b * HID + ug] = newv;
    }
    grid_sync(ctr, NWG * (++sync_round));      // h2q complete across branch
  }
  // T even -> final h2 ends in buffer 0 of each branch.
}

// ---------------------------------------------------------------- head
__global__ void out_init_kernel(const float* __restrict__ out_b, float* __restrict__ out) {
  if (threadIdx.x < BATCH) out[threadIdx.x] = out_b[0];
}

__global__ __launch_bounds__(32, 1)
void head_kernel(const unsigned short* __restrict__ h2L,   // [B][H] bf16 (final)
                 const unsigned short* __restrict__ h2R,
                 const unsigned short* __restrict__ linWb,  // [LIN][2H] bf16
                 const float* __restrict__ lin_b,
                 const float* __restrict__ out_W,
                 float* __restrict__ out) {
  const int n0 = blockIdx.x * 16;
  const int lane = threadIdx.x;
  v8f acc0 = {}; v8f acc1 = {};
  const unsigned short* Wr = linWb + n0 * (2 * HID);
  #pragma unroll 4
  for (int k0 = 0; k0 < HID; k0 += 32) {               // left half (h2L)
    v16bf bf = load_bfrag(Wr, 0, k0, 2 * HID, lane);
    v16bf a0 = load_afrag(h2L, 0,  k0, HID, lane);
    v16bf a1 = load_afrag(h2L, 16, k0, HID, lane);
    acc0 = WMMA_BF16(a0, bf, acc0);
    acc1 = WMMA_BF16(a1, bf, acc1);
  }
  #pragma unroll 4
  for (int k0 = 0; k0 < HID; k0 += 32) {               // right half (h2R)
    v16bf bf = load_bfrag(Wr, 0, HID + k0, 2 * HID, lane);
    v16bf a0 = load_afrag(h2R, 0,  k0, HID, lane);
    v16bf a1 = load_afrag(h2R, 16, k0, HID, lane);
    acc0 = WMMA_BF16(a0, bf, acc0);
    acc1 = WMMA_BF16(a1, bf, acc1);
  }
  __shared__ float st[BATCH][16];
  int nn = lane & 15, mb = (lane >> 4) << 3;
  float lb = lin_b[n0 + nn], ow = out_W[n0 + nn];
  #pragma unroll
  for (int r = 0; r < 8; ++r) {
    st[mb + r][nn]      = fmaxf(acc0[r] + lb, 0.0f) * ow;
    st[16 + mb + r][nn] = fmaxf(acc1[r] + lb, 0.0f) * ow;
  }
  __syncthreads();
  float s = 0.0f;
  #pragma unroll
  for (int j = 0; j < 16; ++j) s += st[lane][j];
  atomicAdd(&out[lane], s);
}

// ---------------------------------------------------------------- launch
extern "C" void kernel_launch(void* const* d_in, const int* in_sizes, int n_in,
                              void* d_out, int out_size, void* d_ws, size_t ws_size,
                              hipStream_t stream) {
  const float* emb   = (const float*)d_in[0];
  const float* W_ih0 = (const float*)d_in[1];
  const float* W_hh0 = (const float*)d_in[2];
  const float* b_ih0 = (const float*)d_in[3];
  const float* b_hh0 = (const float*)d_in[4];
  const float* W_ih1 = (const float*)d_in[5];
  const float* W_hh1 = (const float*)d_in[6];
  const float* b_ih1 = (const float*)d_in[7];
  const float* b_hh1 = (const float*)d_in[8];
  const float* lin_W = (const float*)d_in[9];
  const float* lin_b = (const float*)d_in[10];
  const float* out_W = (const float*)d_in[11];
  const float* out_b = (const float*)d_in[12];
  const int* tokL = (const int*)d_in[13];
  const int* tokR = (const int*)d_in[14];
  const int* lenL = (const int*)d_in[15];
  const int* lenR = (const int*)d_in[16];
  float* out = (float*)d_out;

  // ---- workspace layout (bytes)
  char* ws = (char*)d_ws;
  const size_t stateElems = 2ull * 2 * BATCH * HID;        // [branch][buf][B][H]
  unsigned*       sync_ctr = (unsigned*)ws;                // 256 B (2 counters, padded)
  unsigned short* h1g  = (unsigned short*)(ws + 256);
  unsigned short* h2g  = h1g + stateElems;
  unsigned short* Wih0b = h2g + stateElems;
  unsigned short* Whh0b = Wih0b + (size_t)4 * HID * EMB;
  unsigned short* Wih1b = Whh0b + (size_t)4 * HID * HID;
  unsigned short* Whh1b = Wih1b + (size_t)4 * HID * HID;
  unsigned short* linWb = Whh1b + (size_t)4 * HID * HID;
  // total: 256 + 512KB state + 29MB bf16 weights (ws_size assumed sufficient)

  // zero sync counters + both ping-pong h states every launch (graph-safe)
  hipMemsetAsync(ws, 0, 256 + 2 * stateElems * sizeof(unsigned short), stream);

  // f32 -> bf16 weight conversion
  cvt_bf16_kernel<<<256, 256, 0, stream>>>(W_ih0, Wih0b, 4 * HID * EMB);
  cvt_bf16_kernel<<<512, 256, 0, stream>>>(W_hh0, Whh0b, 4 * HID * HID);
  cvt_bf16_kernel<<<512, 256, 0, stream>>>(W_ih1, Wih1b, 4 * HID * HID);
  cvt_bf16_kernel<<<512, 256, 0, stream>>>(W_hh1, Whh1b, 4 * HID * HID);
  cvt_bf16_kernel<<<512, 256, 0, stream>>>(lin_W, linWb, (size_t)LINW * 2 * HID);

  // persistent recurrent kernel: 2 branches x 64 WGs, 2 grid barriers/step
  lstm_recurrent<<<2 * NWG, THREADS, 0, stream>>>(
      emb, Wih0b, Whh0b, Wih1b, Whh1b,
      b_ih0, b_hh0, b_ih1, b_hh1,
      tokL, tokR, lenL, lenR,
      h1g, h2g, sync_ctr);

  // head: out[b] = sum_j relu(h·linW^T + lin_b)[j] * out_W[j] + out_b
  out_init_kernel<<<1, 32, 0, stream>>>(out_b, out);
  head_kernel<<<LINW / 16, 32, 0, stream>>>(
      h2g /*branch0 buf0*/, h2g + 2 * BATCH * HID /*branch1 buf0*/,
      linWb, lin_b, out_W, out);
}
